// GroupedQueryAttention_8675833938294
// MI455X (gfx1250) — matrix operand order repaired and checked by reading, not compiled, not run on hardware
//
#include <hip/hip_runtime.h>
#include <hip/hip_bf16.h>

typedef __bf16 v16bf __attribute__((ext_vector_type(16)));
typedef float  v8f   __attribute__((ext_vector_type(8)));
typedef int    v4i   __attribute__((ext_vector_type(4)));

#define S_LEN 2048
#define NHEAD 16
#define NKVH  8
#define HDIM  128
#define HID   2048
#define BATCH 2
#define MTOT  (BATCH * S_LEN)   // 4096

#if defined(__HIP_DEVICE_COMPILE__) && __has_builtin(__builtin_amdgcn_global_load_async_to_lds_b128)
#define HAS_ASYNC_LDS 1
#else
#define HAS_ASYNC_LDS 0
#endif

typedef __attribute__((address_space(1))) v4i as1_v4i;
typedef __attribute__((address_space(3))) v4i as3_v4i;

// async (VGPR-bypassing, ASYNCcnt-tracked) global->LDS 16B copy, sync fallback
__device__ __forceinline__ void async_copy_b128(const void* g, void* l) {
#if HAS_ASYNC_LDS
    __builtin_amdgcn_global_load_async_to_lds_b128((as1_v4i*)g, (as3_v4i*)l, 0, 0);
#else
    *reinterpret_cast<uint4*>(l) = *reinterpret_cast<const uint4*>(g);
#endif
}

template <int N>
__device__ __forceinline__ void wait_async() {
#if HAS_ASYNC_LDS
#if __has_builtin(__builtin_amdgcn_s_wait_asynccnt)
    __builtin_amdgcn_s_wait_asynccnt(N);
#else
    asm volatile("s_wait_asynccnt %0" ::"n"(N) : "memory");
#endif
#endif
}

union FragU { uint4 u[2]; v16bf v; };

__device__ __forceinline__ v16bf load_frag(const __bf16* p0, const __bf16* p1) {
    FragU f;
    f.u[0] = *reinterpret_cast<const uint4*>(p0);
    f.u[1] = *reinterpret_cast<const uint4*>(p1);
    return f.v;
}

__device__ __forceinline__ v8f wmma_bf16(v16bf a, v16bf b, v8f c) {
    return __builtin_amdgcn_wmma_f32_16x16x32_bf16(false, a, false, b, (short)0, c, false, false);
}

// ---------------------------------------------------------------- fp32 -> bf16
__global__ void cvt_f32_bf16(const float* __restrict__ in, __bf16* __restrict__ out, int n4) {
    int i = blockIdx.x * blockDim.x + threadIdx.x;
    if (i >= n4) return;
    float4 f = reinterpret_cast<const float4*>(in)[i];
    union { __bf16 h[4]; uint2 u; } t;
    t.h[0] = (__bf16)f.x; t.h[1] = (__bf16)f.y;
    t.h[2] = (__bf16)f.z; t.h[3] = (__bf16)f.w;
    reinterpret_cast<uint2*>(out)[i] = t.u;
}

// ---------------------------------------------------------------- WMMA GEMM
// C[M=4096][N] = A[M][K] * W[N][K]^T   (A,W bf16, accum fp32)
// mode 0: fp32 out, layout [B, nheads, S, HD]
// mode 1: bf16 out, layout [B, nheads, S, HD]
// mode 2: fp32 out, row-major [M][N]
__launch_bounds__(256)
__global__ void gemm_wmma(const __bf16* __restrict__ A, const __bf16* __restrict__ W,
                          float* __restrict__ outf, __bf16* __restrict__ outb,
                          int N, int K, int mode, int nheads) {
    __shared__ __align__(16) __bf16 As[2][128][40];
    __shared__ __align__(16) __bf16 Ws[2][128][40];
    const int tid  = threadIdx.x;
    const int lane = tid & 31, wave = tid >> 5;
    const int wm = wave >> 2, wn = wave & 3;
    const int m0 = blockIdx.y * 128, n0 = blockIdx.x * 128;
    const int ml  = lane & 15;
    const int kbA = (lane < 16) ? 0 : 8;    // A-fragment K base
    const int kbB = (lane < 16) ? 0 : 16;   // unused; B fragment is loaded with kbA

    // each thread stages rows r0 and r0+64, 8 bf16 at column c0, for both A and W
    const int r0 = tid >> 2;
    const int c0 = (tid & 3) << 3;

    const v8f vzero = {0.f,0.f,0.f,0.f,0.f,0.f,0.f,0.f};
    v8f acc[4][2];
    #pragma unroll
    for (int i = 0; i < 4; ++i)
        #pragma unroll
        for (int j = 0; j < 2; ++j) acc[i][j] = vzero;

    auto stage = [&](int k0, int buf) {
        async_copy_b128(&A[(size_t)(m0 + r0) * K + k0 + c0],      &As[buf][r0][c0]);
        async_copy_b128(&A[(size_t)(m0 + r0 + 64) * K + k0 + c0], &As[buf][r0 + 64][c0]);
        async_copy_b128(&W[(size_t)(n0 + r0) * K + k0 + c0],      &Ws[buf][r0][c0]);
        async_copy_b128(&W[(size_t)(n0 + r0 + 64) * K + k0 + c0], &Ws[buf][r0 + 64][c0]);
    };

    stage(0, 0);
    int cur = 0;
    for (int k0 = 0; k0 < K; k0 += 32, cur ^= 1) {
        if (k0 + 32 < K) {
            stage(k0 + 32, cur ^ 1);   // prefetch next tile into other buffer
            wait_async<4>();           // current tile's 4 async ops complete (in-order)
        } else {
            wait_async<0>();
        }
        __syncthreads();

        v16bf af[4], bf_[2];
        #pragma unroll
        for (int mi = 0; mi < 4; ++mi) {
            const __bf16* rp = &As[cur][wm * 64 + mi * 16 + ml][0];
            af[mi] = load_frag(rp + kbA, rp + kbA + 16);
        }
        #pragma unroll
        for (int ni = 0; ni < 2; ++ni) {
            const __bf16* rp = &Ws[cur][wn * 32 + ni * 16 + ml][0];
            bf_[ni] = load_frag(rp + kbA, rp + kbA + 16);
        }
        #pragma unroll
        for (int mi = 0; mi < 4; ++mi)
            #pragma unroll
            for (int ni = 0; ni < 2; ++ni)
                acc[mi][ni] = wmma_bf16(af[mi], bf_[ni], acc[mi][ni]);
        __syncthreads();
    }

    const int mh = (lane >> 4) << 3;   // 0 or 8
    #pragma unroll
    for (int mi = 0; mi < 4; ++mi)
        #pragma unroll
        for (int ni = 0; ni < 2; ++ni)
            #pragma unroll
            for (int v = 0; v < 8; ++v) {
                int gm = m0 + wm * 64 + mi * 16 + mh + v;
                int gn = n0 + wn * 32 + ni * 16 + ml;
                float val = acc[mi][ni][v];
                if (mode == 2) {
                    outf[(size_t)gm * N + gn] = val;
                } else {
                    int bb = gm >> 11, s = gm & (S_LEN - 1);
                    int head = gn >> 7, d = gn & (HDIM - 1);
                    size_t idx = (((size_t)bb * nheads + head) * S_LEN + s) * HDIM + d;
                    if (mode == 0) outf[idx] = val;
                    else           outb[idx] = (__bf16)val;
                }
            }
}

// ---------------------------------------------------------------- RoPE (+scale) fp32 -> bf16
__global__ void rope_kernel(const float* __restrict__ in, __bf16* __restrict__ out,
                            float scale, int total) {
    int idx = blockIdx.x * blockDim.x + threadIdx.x;
    if (idx >= total) return;
    int d   = idx & 63;
    int tmp = idx >> 6;            // bh * S + s
    int s   = tmp & (S_LEN - 1);
    size_t base = (size_t)tmp * HDIM;
    float x0 = in[base + d], x1 = in[base + d + 64];
    // theta = s * 10000^(-d/64)  ;  ln(10000)/64 = 0.14391156...
    float th = (float)s * __expf(-(float)d * 0.14391156f);
    float sn, cs;
    __sincosf(th, &sn, &cs);
    out[base + d]      = (__bf16)((x0 * cs - x1 * sn) * scale);
    out[base + 64 + d] = (__bf16)((x1 * cs + x0 * sn) * scale);
}

// ---------------------------------------------------------------- flash attention (causal, GQA)
// Q: [B,NH,S,HD] bf16 (pre-scaled by 1/sqrt(HD)), K/V: [B,NKV,S,HD] bf16
// O: [B,S,NH*HD] bf16
__launch_bounds__(128)
__global__ void flash_attn(const __bf16* __restrict__ Q, const __bf16* __restrict__ Kt,
                           const __bf16* __restrict__ V, __bf16* __restrict__ O) {
    __shared__ __align__(16) __bf16 Ks[32][136];     // [kv][hd], padded
    __shared__ __align__(16) __bf16 Vt[128][40];     // [hd][kv], padded
    __shared__ __align__(16) __bf16 Pl[4][16][40];   // per-wave P relayout buffer

    const int tid = threadIdx.x, lane = tid & 31, wave = tid >> 5;
    const int b = blockIdx.z, h = blockIdx.y;
    const int qb0 = blockIdx.x * 64;
    const int q0w = qb0 + wave * 16;
    const int ml  = lane & 15;
    const int kbA = (lane < 16) ? 0 : 8;
    const int kbB = (lane < 16) ? 0 : 16;
    const int rl  = (lane < 16) ? 0 : 8;

    const __bf16* qp = Q  + ((size_t)(b * NHEAD + h)        * S_LEN) * HDIM;
    const __bf16* kp = Kt + ((size_t)(b * NKVH  + (h >> 1)) * S_LEN) * HDIM;
    const __bf16* vp = V  + ((size_t)(b * NKVH  + (h >> 1)) * S_LEN) * HDIM;

    v16bf qf[4];
    #pragma unroll
    for (int c = 0; c < 4; ++c) {
        const __bf16* rp = qp + (size_t)(q0w + ml) * HDIM + c * 32;
        qf[c] = load_frag(rp + kbA, rp + kbA + 16);
    }

    const v8f vzero = {0.f,0.f,0.f,0.f,0.f,0.f,0.f,0.f};
    v8f oacc[8];
    float mrow[8], lrow[8];
    #pragma unroll
    for (int t = 0; t < 8; ++t) oacc[t] = vzero;
    #pragma unroll
    for (int v = 0; v < 8; ++v) { mrow[v] = -1e38f; lrow[v] = 0.f; }

    for (int kv0 = 0; kv0 < qb0 + 64; kv0 += 32) {
        // ---- staging: K async global->LDS; V batched loads then LDS transpose ----
        uint4 vt4[4];
        #pragma unroll
        for (int i = 0; i < 4; ++i) {
            int v  = tid + i * 128;
            int r  = v >> 4;
            int c8 = (v & 15) << 3;
            async_copy_b128(&kp[(size_t)(kv0 + r) * HDIM + c8], &Ks[r][c8]);
            vt4[i] = *reinterpret_cast<const uint4*>(&vp[(size_t)(kv0 + r) * HDIM + c8]);
        }
        #pragma unroll
        for (int i = 0; i < 4; ++i) {
            int v  = tid + i * 128;
            int r  = v >> 4;
            int c8 = (v & 15) << 3;
            union { uint4 u; __bf16 h[8]; } t;
            t.u = vt4[i];
            #pragma unroll
            for (int j = 0; j < 8; ++j) Vt[c8 + j][r] = t.h[j];
        }
        wait_async<0>();
        __syncthreads();

        if (kv0 <= q0w + 15) {   // wave-uniform: block not fully masked
            v8f s0 = vzero, s1 = vzero;
            #pragma unroll
            for (int c = 0; c < 4; ++c) {
                const __bf16* r0 = &Ks[ml][c * 32];
                const __bf16* r1 = &Ks[16 + ml][c * 32];
                v16bf k0f = load_frag(r0 + kbA, r0 + kbA + 16);
                v16bf k1f = load_frag(r1 + kbA, r1 + kbA + 16);
                s0 = wmma_bf16(qf[c], k0f, s0);
                s1 = wmma_bf16(qf[c], k1f, s1);
            }
            if (kv0 + 32 > q0w) {   // diagonal block: per-element causal mask
                #pragma unroll
                for (int v = 0; v < 8; ++v) {
                    int row = q0w + rl + v;
                    if (kv0 + ml > row)      s0[v] = -1e38f;
                    if (kv0 + 16 + ml > row) s1[v] = -1e38f;
                }
            }
            float a8[8];
            #pragma unroll
            for (int v = 0; v < 8; ++v) {
                float rm = fmaxf(s0[v], s1[v]);
                #pragma unroll
                for (int off = 1; off < 16; off <<= 1)
                    rm = fmaxf(rm, __shfl_xor(rm, off, 32));
                float mo = mrow[v];
                float mn = fmaxf(mo, rm);
                mrow[v] = mn;
                float al = __expf(mo - mn);
                a8[v] = al;
                lrow[v] *= al;
                float p0 = (s0[v] < -1e30f) ? 0.f : __expf(s0[v] - mn);
                float p1 = (s1[v] < -1e30f) ? 0.f : __expf(s1[v] - mn);
                float rs = p0 + p1;
                #pragma unroll
                for (int off = 1; off < 16; off <<= 1)
                    rs += __shfl_xor(rs, off, 32);
                lrow[v] += rs;
                Pl[wave][rl + v][ml]      = (__bf16)p0;
                Pl[wave][rl + v][16 + ml] = (__bf16)p1;
            }
            #pragma unroll
            for (int t = 0; t < 8; ++t)
                #pragma unroll
                for (int v = 0; v < 8; ++v) oacc[t][v] *= a8[v];
            // P: C-layout -> A-fragment via per-wave LDS (same-wave DS ops are in order)
            const __bf16* pp = &Pl[wave][ml][0];
            v16bf pf = load_frag(pp + kbA, pp + kbA + 16);
            #pragma unroll
            for (int t = 0; t < 8; ++t) {
                const __bf16* vr = &Vt[t * 16 + ml][0];
                v16bf vf = load_frag(vr + kbA, vr + kbA + 16);
                oacc[t] = wmma_bf16(pf, vf, oacc[t]);
            }
        }
        __syncthreads();
    }

    float invl[8];
    #pragma unroll
    for (int v = 0; v < 8; ++v) invl[v] = 1.f / lrow[v];
    #pragma unroll
    for (int t = 0; t < 8; ++t)
        #pragma unroll
        for (int v = 0; v < 8; ++v) {
            int row = q0w + rl + v;
            int col = h * HDIM + t * 16 + ml;
            O[((size_t)b * S_LEN + row) * HID + col] = (__bf16)(oacc[t][v] * invl[v]);
        }
}

// ---------------------------------------------------------------- host launch
extern "C" void kernel_launch(void* const* d_in, const int* in_sizes, int n_in,
                              void* d_out, int out_size, void* d_ws, size_t ws_size,
                              hipStream_t stream) {
    (void)in_sizes; (void)n_in; (void)out_size; (void)ws_size;
    const float* hs = (const float*)d_in[0];
    // d_in[1] = attention_mask (pure causal; reproduced analytically)
    const float* wq = (const float*)d_in[2];
    const float* wk = (const float*)d_in[3];
    const float* wv = (const float*)d_in[4];
    const float* wo = (const float*)d_in[5];
    float* out = (float*)d_out;

    char* p = (char*)d_ws;
    auto carve = [&](size_t bytes) {
        void* r = (void*)p;
        p += (bytes + 255) & ~(size_t)255;
        return r;
    };
    __bf16* hbf = (__bf16*)carve((size_t)MTOT * HID * 2);           // 16 MB
    __bf16* wqb = (__bf16*)carve((size_t)HID * HID * 2);            //  8 MB
    __bf16* wkb = (__bf16*)carve((size_t)(NKVH*HDIM) * HID * 2);    //  4 MB
    __bf16* wvb = (__bf16*)carve((size_t)(NKVH*HDIM) * HID * 2);    //  4 MB
    __bf16* wob = (__bf16*)carve((size_t)HID * HID * 2);            //  8 MB
    float*  qfp = (float*) carve((size_t)BATCH*NHEAD*S_LEN*HDIM*4); // 32 MB
    float*  kfp = (float*) carve((size_t)BATCH*NKVH *S_LEN*HDIM*4); // 16 MB
    __bf16* qbb = (__bf16*)carve((size_t)BATCH*NHEAD*S_LEN*HDIM*2); // 16 MB
    __bf16* kbb = (__bf16*)carve((size_t)BATCH*NKVH *S_LEN*HDIM*2); //  8 MB
    __bf16* vbb = (__bf16*)carve((size_t)BATCH*NKVH *S_LEN*HDIM*2); //  8 MB
    __bf16* obb = (__bf16*)qfp;  // reuse qfp region (needed: 16 MB <= 32 MB; qfp dead post-RoPE)

    // 1) fp32 -> bf16 conversions
    cvt_f32_bf16<<<(MTOT*HID/4 + 255)/256, 256, 0, stream>>>(hs, hbf, MTOT*HID/4);
    cvt_f32_bf16<<<(HID*HID/4  + 255)/256, 256, 0, stream>>>(wq, wqb, HID*HID/4);
    cvt_f32_bf16<<<(NKVH*HDIM*HID/4 + 255)/256, 256, 0, stream>>>(wk, wkb, NKVH*HDIM*HID/4);
    cvt_f32_bf16<<<(NKVH*HDIM*HID/4 + 255)/256, 256, 0, stream>>>(wv, wvb, NKVH*HDIM*HID/4);
    cvt_f32_bf16<<<(HID*HID/4  + 255)/256, 256, 0, stream>>>(wo, wob, HID*HID/4);

    // 2) projections: Q (fp32), K (fp32), V (bf16 direct)
    gemm_wmma<<<dim3(HID/128,        MTOT/128), 256, 0, stream>>>(hbf, wqb, qfp, nullptr, HID,       HID, 0, NHEAD);
    gemm_wmma<<<dim3(NKVH*HDIM/128,  MTOT/128), 256, 0, stream>>>(hbf, wkb, kfp, nullptr, NKVH*HDIM, HID, 0, NKVH);
    gemm_wmma<<<dim3(NKVH*HDIM/128,  MTOT/128), 256, 0, stream>>>(hbf, wvb, nullptr, vbb, NKVH*HDIM, HID, 1, NKVH);

    // 3) RoPE (+ fold 1/sqrt(HD) into Q) -> bf16
    const float qscale = 0.08838834764831845f;  // 1/sqrt(128)
    int totq = BATCH * NHEAD * S_LEN * (HDIM/2);
    int totk = BATCH * NKVH  * S_LEN * (HDIM/2);
    rope_kernel<<<totq/256, 256, 0, stream>>>(qfp, qbb, qscale, totq);
    rope_kernel<<<totk/256, 256, 0, stream>>>(kfp, kbb, 1.0f,   totk);

    // 4) causal flash attention -> [B,S,NH*HD] bf16
    flash_attn<<<dim3(S_LEN/64, NHEAD, BATCH), 128, 0, stream>>>(qbb, kbb, vbb, obb);

    // 5) output projection -> fp32 d_out
    gemm_wmma<<<dim3(HID/128, MTOT/128), 256, 0, stream>>>(obb, wob, out, nullptr, HID, HID, 2, NHEAD);
}
